// TreeDensityTransformer_80625126080567
// MI455X (gfx1250) — compile-verified, matrix operand order checked by reading, not run on hardware
//
#include <hip/hip_runtime.h>

// ---------------------------------------------------------------------------
// TreeDensityTransformer for MI455X (gfx1250, wave32, WMMA)
// GEMMs: v_wmma_f32_16x16x32_bf16, fp32 accum. Weights pre-transposed to
// column-major bf16 so A and B LDS fragment gathers are both K-contiguous
// (fuse into ds_load_b128). LDS tiles are double-buffered and staged with
// GLOBAL_LOAD_ASYNC_TO_LDS_B128 (ASYNCcnt) when the toolchain exposes it.
// ---------------------------------------------------------------------------

typedef __bf16 bf16_t;
typedef __attribute__((ext_vector_type(16))) __bf16 v16bf;
typedef __attribute__((ext_vector_type(2)))  __bf16 v2bf;
typedef __attribute__((ext_vector_type(8)))  float  v8f;
typedef __attribute__((ext_vector_type(4)))  int    v4i;

#define B_    16
#define L_    2048
#define D_    512
#define H_    8
#define DH_   64
#define F_    2048
#define NL_   4
#define V_    21
#define MROWS (B_ * L_)          // 32768 token rows

#if defined(__HIP_DEVICE_COMPILE__) && __has_builtin(__builtin_amdgcn_global_load_async_to_lds_b128)
#define USE_ASYNC_LDS 1
#else
#define USE_ASYNC_LDS 0
#endif

// ---------------- helpers ----------------

__device__ __forceinline__ bf16_t f2bf(float f) {
  union { float f; unsigned u; } c; c.f = f;
  unsigned u = c.u;
  u += 0x7FFFu + ((u >> 16) & 1u);     // round-to-nearest-even
  union { unsigned short s; bf16_t b; } o; o.s = (unsigned short)(u >> 16);
  return o.b;
}

__device__ __forceinline__ float bf2f(bf16_t b) {
  union { unsigned short s; bf16_t b; } i; i.b = b;
  union { unsigned u; float f; } o; o.u = ((unsigned)i.s) << 16;
  return o.f;
}

__device__ __forceinline__ float wave_sum32(float x) {
#pragma unroll
  for (int m = 16; m >= 1; m >>= 1) x += __shfl_xor(x, m, 32);
  return x;
}

// 16-byte global -> LDS copy: async (ASYNCcnt) when available, sync fallback.
__device__ __forceinline__ void copy16_g2l(const bf16_t* g, bf16_t* l) {
#if USE_ASYNC_LDS
  __builtin_amdgcn_global_load_async_to_lds_b128(
      (__attribute__((address_space(1))) v4i*)g,
      (__attribute__((address_space(3))) v4i*)l, 0, 0);
#else
  *(uint4*)l = *(const uint4*)g;
#endif
}

__device__ __forceinline__ void async_wait0() {
#if USE_ASYNC_LDS
#if __has_builtin(__builtin_amdgcn_s_wait_asynccnt)
  __builtin_amdgcn_s_wait_asynccnt(0);
#else
  asm volatile("s_wait_asynccnt 0x0" ::: "memory");
#endif
#endif
}

// ---------------- embedding: h = tok[x] + pos ----------------

__global__ void embed_kernel(const int* __restrict__ x,
                             const float* __restrict__ tok,
                             const float* __restrict__ pos,
                             float* __restrict__ h, bf16_t* __restrict__ hb) {
  size_t i = (size_t)blockIdx.x * blockDim.x + threadIdx.x;
  if (i >= (size_t)MROWS * D_) return;
  int d = (int)(i & (D_ - 1));
  size_t bl = i >> 9;                       // D_ = 512
  int l = (int)(bl & (L_ - 1));
  float v = tok[(size_t)x[bl] * D_ + d] + pos[(size_t)l * D_ + d];
  h[i]  = v;
  hb[i] = f2bf(v);
}

// ---------------- fp32 [K][N] -> bf16 [N][K] transposing weight convert ------

__global__ void cvt_t_kernel(const float* __restrict__ in, bf16_t* __restrict__ out,
                             int K, int N) {
  int i = blockIdx.x * blockDim.x + threadIdx.x;
  if (i >= K * N) return;
  int k = i / N, n = i - k * N;
  out[(size_t)n * K + k] = f2bf(in[i]);
}

// ---------------- WMMA bf16 GEMM ----------------
// C[M,N] = A[M,K] * Bt[N,K]^T + bias ; block tile 128x256x32, 8 waves,
// each wave computes a 64x64 patch = 4x4 WMMA tiles (16 accumulators).
// Double-buffered LDS staging overlaps async copies with WMMA compute.

#define TM 128
#define TN 256
#define TK 32
#define LDT (TK + 8)

// Gather a 16x32 bf16 fragment from an LDS tile stored row(=M or N)-major with
// K contiguous. CDNA5 16-bit layout: lane -> row (lane&15), element e -> K =
// e + (e>=8?8:0) + (lane>=16?8:0).
__device__ __forceinline__ v16bf load_frag(const bf16_t* __restrict__ base,
                                           int lm, int hi) {
  v16bf f;
  const bf16_t* p = base + lm * LDT;
#pragma unroll
  for (int j = 0; j < 8; ++j) {
    const int e  = j * 2;
    const int kk = e + ((e >= 8) ? 8 : 0) + hi * 8;
    v2bf t = *(const v2bf*)(p + kk);
    f[e]     = t[0];
    f[e + 1] = t[1];
  }
  return f;
}

template <int RELU, int OUTF, int OUTB>
__global__ __launch_bounds__(256)
void gemm_bf16_wmma(const bf16_t* __restrict__ A, const bf16_t* __restrict__ Bt,
                    const float* __restrict__ bias,
                    float* __restrict__ Cf, bf16_t* __restrict__ Cb,
                    int Ndim, int Kdim) {
  __shared__ bf16_t As[2][TM][LDT];   // 2 x 10.0 KB
  __shared__ bf16_t Bs[2][TN][LDT];   // 2 x 20.0 KB   (total ~60 KB of 320 KB)

  const int tid  = threadIdx.x;
  const int lane = tid & 31;
  const int wave = tid >> 5;
  const int hi   = lane >> 4;     // K-half selector for 16-bit fragments
  const int lm   = lane & 15;

  const int rowBase = blockIdx.y * TM;
  const int colBase = blockIdx.x * TN;
  const int wr = (wave >> 2) * 64;   // 0 / 64        (2 wave-rows)
  const int wc = (wave & 3) * 64;    // 0/64/128/192  (4 wave-cols)

  // stage one K-tile (A: 512 16B chunks, B: 1024 16B chunks; 256 threads)
  auto stage = [&](int buf, int kb) {
#pragma unroll
    for (int it = 0; it < 2; ++it) {
      int idx = tid + it * 256;          // 0..511
      int r   = idx >> 2;                // 0..127
      int cg  = (idx & 3) << 3;          // 0,8,16,24
      copy16_g2l(A + (size_t)(rowBase + r) * Kdim + kb + cg, &As[buf][r][cg]);
    }
#pragma unroll
    for (int it = 0; it < 4; ++it) {
      int idx = tid + it * 256;          // 0..1023
      int r   = idx >> 2;                // 0..255
      int cg  = (idx & 3) << 3;
      copy16_g2l(Bt + (size_t)(colBase + r) * Kdim + kb + cg, &Bs[buf][r][cg]);
    }
  };

  v8f acc[4][4];
#pragma unroll
  for (int mi = 0; mi < 4; ++mi)
#pragma unroll
    for (int ni = 0; ni < 4; ++ni)
#pragma unroll
      for (int e = 0; e < 8; ++e) acc[mi][ni][e] = 0.0f;

  stage(0, 0);
  async_wait0();
  __syncthreads();

  int buf = 0;
  for (int kb = 0; kb < Kdim; kb += TK) {
    const int nk = kb + TK;
    if (nk < Kdim) stage(buf ^ 1, nk);   // stream next tile while computing

    v16bf af[4];
#pragma unroll
    for (int mi = 0; mi < 4; ++mi)
      af[mi] = load_frag(&As[buf][wr + mi * 16][0], lm, hi);

#pragma unroll
    for (int ni = 0; ni < 4; ++ni) {
      v16bf bfr = load_frag(&Bs[buf][wc + ni * 16][0], lm, hi);
#pragma unroll
      for (int mi = 0; mi < 4; ++mi)
        acc[mi][ni] = __builtin_amdgcn_wmma_f32_16x16x32_bf16(
            false, af[mi], false, bfr, (short)0, acc[mi][ni], false, false);
    }

    async_wait0();      // next-tile LDS writes landed
    __syncthreads();    // everyone done reading current tile
    buf ^= 1;
  }

  // epilogue: hoisted bias, compile-time RELU / output-format selection.
  float bias_r[4];
#pragma unroll
  for (int ni = 0; ni < 4; ++ni)
    bias_r[ni] = bias[colBase + wc + ni * 16 + lm];

#pragma unroll
  for (int mi = 0; mi < 4; ++mi)
#pragma unroll
    for (int ni = 0; ni < 4; ++ni) {
      size_t o = (size_t)(rowBase + wr + mi * 16 + hi * 8) * Ndim
               + (colBase + wc + ni * 16 + lm);
#pragma unroll
      for (int r = 0; r < 8; ++r) {
        float v = acc[mi][ni][r] + bias_r[ni];
        if (RELU) v = fmaxf(v, 0.0f);
        if (OUTF) Cf[o] = v;
        if (OUTB) Cb[o] = f2bf(v);
        o += (size_t)Ndim;
      }
    }
}

// ---------------- tree attention: 2 keys (self, parent), wave32 per (b,l,h) --

__global__ void attn_kernel(const bf16_t* __restrict__ q, const bf16_t* __restrict__ k,
                            const bf16_t* __restrict__ v, const int* __restrict__ parents,
                            bf16_t* __restrict__ o) {
  int gid = blockIdx.x * (blockDim.x >> 5) + (threadIdx.x >> 5);
  if (gid >= MROWS * H_) return;
  int h  = gid & (H_ - 1);
  int bl = gid >> 3;                 // H_ = 8
  int l  = bl & (L_ - 1);
  int b  = bl >> 11;                 // L_ = 2048
  int lane = threadIdx.x & 31;

  size_t base = (size_t)bl * D_ + h * DH_;
  int p  = parents[l];
  int pk = (p < 0) ? l : (p + 1);
  size_t pbase = ((size_t)(b * L_ + pk)) * D_ + h * DH_;

  float q0 = bf2f(q[base + lane]),  q1 = bf2f(q[base + 32 + lane]);
  float k0 = bf2f(k[base + lane]),  k1 = bf2f(k[base + 32 + lane]);
  float p0 = bf2f(k[pbase + lane]), p1 = bf2f(k[pbase + 32 + lane]);

  float s0 = wave_sum32(q0 * k0 + q1 * k1) * 0.125f;   // 1/sqrt(64)
  float s1 = wave_sum32(q0 * p0 + q1 * p1) * 0.125f;
  if (p < 0) s1 = -1e9f;

  float mx = fmaxf(s0, s1);
  float e0 = __expf(s0 - mx), e1 = __expf(s1 - mx);
  float inv = 1.0f / (e0 + e1);
  float w0 = e0 * inv, w1 = e1 * inv;

  float o0 = w0 * bf2f(v[base + lane])      + w1 * bf2f(v[pbase + lane]);
  float o1 = w0 * bf2f(v[base + 32 + lane]) + w1 * bf2f(v[pbase + 32 + lane]);
  o[base + lane]      = f2bf(o0);
  o[base + 32 + lane] = f2bf(o1);
}

// ---------------- fused residual + LayerNorm (in-place on h), wave32 per row -

__global__ void add_ln_kernel(float* __restrict__ h, const float* __restrict__ resid,
                              const float* __restrict__ s, const float* __restrict__ b,
                              bf16_t* __restrict__ hb) {
  int row = blockIdx.x * (blockDim.x >> 5) + (threadIdx.x >> 5);
  if (row >= MROWS) return;
  int lane = threadIdx.x & 31;
  size_t base = (size_t)row * D_;

  float x[16];
  float sum = 0.0f;
#pragma unroll
  for (int j = 0; j < 16; ++j) {
    int c = j * 32 + lane;
    x[j] = h[base + c] + resid[base + c];
    sum += x[j];
  }
  float mu = wave_sum32(sum) * (1.0f / D_);
  float vs = 0.0f;
#pragma unroll
  for (int j = 0; j < 16; ++j) { float d = x[j] - mu; vs += d * d; }
  float var = wave_sum32(vs) * (1.0f / D_);
  float rs  = rsqrtf(var + 1e-6f);
#pragma unroll
  for (int j = 0; j < 16; ++j) {
    int c = j * 32 + lane;
    float y = (x[j] - mu) * rs * s[c] + b[c];
    h[base + c]  = y;
    hb[base + c] = f2bf(y);
  }
}

// ---------------- output projection: [32768,512] x [512,21] + bout ----------

__global__ void out_kernel(const float* __restrict__ h, const float* __restrict__ Wout,
                           const float* __restrict__ bout, float* __restrict__ out) {
  int row = blockIdx.x * (blockDim.x >> 5) + (threadIdx.x >> 5);
  if (row >= MROWS) return;
  int lane = threadIdx.x & 31;
  size_t base = (size_t)row * D_;
  float hx[16];
#pragma unroll
  for (int j = 0; j < 16; ++j) hx[j] = h[base + j * 32 + lane];
  for (int vv = 0; vv < V_; ++vv) {
    float p = 0.0f;
#pragma unroll
    for (int j = 0; j < 16; ++j) p += hx[j] * Wout[(size_t)(j * 32 + lane) * V_ + vv];
    p = wave_sum32(p);
    if (lane == 0) out[(size_t)row * V_ + vv] = p + bout[vv];
  }
}

// ---------------- orchestration ----------------

static inline void launch_gemm(const bf16_t* A, const bf16_t* Bt, const float* bias,
                               float* Cf, bf16_t* Cb, int N, int K, int relu,
                               hipStream_t stream) {
  dim3 grid(N / TN, MROWS / TM);
  if (Cf)
    gemm_bf16_wmma<0, 1, 0><<<grid, 256, 0, stream>>>(A, Bt, bias, Cf, Cb, N, K);
  else if (relu)
    gemm_bf16_wmma<1, 0, 1><<<grid, 256, 0, stream>>>(A, Bt, bias, Cf, Cb, N, K);
  else
    gemm_bf16_wmma<0, 0, 1><<<grid, 256, 0, stream>>>(A, Bt, bias, Cf, Cb, N, K);
}

extern "C" void kernel_launch(void* const* d_in, const int* in_sizes, int n_in,
                              void* d_out, int out_size, void* d_ws, size_t ws_size,
                              hipStream_t stream) {
  const int*   x     = (const int*)  d_in[0];
  const int*   par   = (const int*)  d_in[1];
  const float* tok   = (const float*)d_in[2];
  const float* pos   = (const float*)d_in[3];
  const float* Wq    = (const float*)d_in[4];
  const float* bq    = (const float*)d_in[5];
  const float* Wk    = (const float*)d_in[6];
  const float* bk    = (const float*)d_in[7];
  const float* Wv    = (const float*)d_in[8];
  const float* bv    = (const float*)d_in[9];
  const float* Wo    = (const float*)d_in[10];
  const float* bo    = (const float*)d_in[11];
  const float* ln1s  = (const float*)d_in[12];
  const float* ln1b  = (const float*)d_in[13];
  const float* ln2s  = (const float*)d_in[14];
  const float* ln2b  = (const float*)d_in[15];
  const float* W1    = (const float*)d_in[16];
  const float* b1    = (const float*)d_in[17];
  const float* W2    = (const float*)d_in[18];
  const float* b2    = (const float*)d_in[19];
  const float* Wout  = (const float*)d_in[20];
  const float* bout  = (const float*)d_in[21];
  float* out = (float*)d_out;

  char* ws = (char*)d_ws;
  const size_t MB = 1ull << 20;
  float*  h    = (float*) (ws + 0);          //  64 MB fp32 residual stream
  bf16_t* hb   = (bf16_t*)(ws + 64  * MB);   //  32 MB bf16 shadow of h
  bf16_t* qb   = (bf16_t*)(ws + 96  * MB);   //  32 MB
  bf16_t* kb2  = (bf16_t*)(ws + 128 * MB);   //  32 MB
  bf16_t* vb2  = (bf16_t*)(ws + 160 * MB);   //  32 MB
  bf16_t* ff1b = (bf16_t*)(ws + 96  * MB);   // 128 MB (reuses q/k/v after attention)
  float*  tmp  = (float*) (ws + 224 * MB);   //  64 MB GEMM fp32 output
  bf16_t* ob   = (bf16_t*)(ws + 288 * MB);   //  32 MB attention output
  bf16_t* wqb  = (bf16_t*)(ws + 320 * MB);   // bf16 transposed weights (~24 MB)
  bf16_t* wkb  = wqb + (size_t)NL_ * D_ * D_;
  bf16_t* wvb  = wkb + (size_t)NL_ * D_ * D_;
  bf16_t* wob  = wvb + (size_t)NL_ * D_ * D_;
  bf16_t* w1b  = wob + (size_t)NL_ * D_ * D_;
  bf16_t* w2b  = w1b + (size_t)NL_ * D_ * F_;

  // transposing weight downconverts: fp32 [K][N] -> bf16 [N][K] per layer
  auto cvt_t = [&](const float* src, bf16_t* dst, int K, int N, size_t lstride) {
    for (int l = 0; l < NL_; ++l)
      cvt_t_kernel<<<(K * N + 255) / 256, 256, 0, stream>>>(
          src + l * lstride, dst + l * lstride, K, N);
  };
  cvt_t(Wq, wqb, D_, D_, (size_t)D_ * D_);
  cvt_t(Wk, wkb, D_, D_, (size_t)D_ * D_);
  cvt_t(Wv, wvb, D_, D_, (size_t)D_ * D_);
  cvt_t(Wo, wob, D_, D_, (size_t)D_ * D_);
  cvt_t(W1, w1b, D_, F_, (size_t)D_ * F_);
  cvt_t(W2, w2b, F_, D_, (size_t)F_ * D_);

  // embedding
  {
    size_t n = (size_t)MROWS * D_;
    embed_kernel<<<(unsigned)((n + 255) / 256), 256, 0, stream>>>(x, tok, pos, h, hb);
  }

  const int attnBlocks = (MROWS * H_) / 8;   // 8 waves per 256-thread block
  const int rowBlocks  = MROWS / 8;

  for (int l = 0; l < NL_; ++l) {
    const size_t wDD = (size_t)l * D_ * D_;
    const size_t wDF = (size_t)l * D_ * F_;
    // QKV projections (bf16 outputs feed attention)
    launch_gemm(hb, wqb + wDD, bq + l * D_, nullptr, qb,  D_, D_, 0, stream);
    launch_gemm(hb, wkb + wDD, bk + l * D_, nullptr, kb2, D_, D_, 0, stream);
    launch_gemm(hb, wvb + wDD, bv + l * D_, nullptr, vb2, D_, D_, 0, stream);
    // 2-key tree attention
    attn_kernel<<<attnBlocks, 256, 0, stream>>>(qb, kb2, vb2, par, ob);
    // output projection
    launch_gemm(ob, wob + wDD, bo + l * D_, tmp, nullptr, D_, D_, 0, stream);
    // residual + LN1 (in-place h, refresh bf16 shadow)
    add_ln_kernel<<<rowBlocks, 256, 0, stream>>>(h, tmp, ln1s + l * D_, ln1b + l * D_, hb);
    // FFN: GEMM1 (fused bias+ReLU, bf16 out), GEMM2 (fp32 out)
    launch_gemm(hb, w1b + wDF, b1 + l * F_, nullptr, ff1b, F_, D_, 1, stream);
    launch_gemm(ff1b, w2b + wDF, b2 + l * D_, tmp, nullptr, D_, F_, 0, stream);
    // residual + LN2
    add_ln_kernel<<<rowBlocks, 256, 0, stream>>>(h, tmp, ln2s + l * D_, ln2b + l * D_, hb);
  }

  // final vocab projection
  out_kernel<<<rowBlocks, 256, 0, stream>>>(h, Wout, bout, out);

  (void)in_sizes; (void)n_in; (void)out_size; (void)ws_size;
}
